// MyNet_11879879542564
// MI455X (gfx1250) — matrix operand order repaired and checked by reading, not compiled, not run on hardware
//
#include <hip/hip_runtime.h>
#include <hip/hip_bf16.h>
#include <stddef.h>

// Problem constants (match reference): B=32, T=512, N_IN=64, H=512, 4H=2048
#define BB 32
#define TT 512
#define NIN 64
#define HH 512
#define G4 2048
#define ARS (TT*HH)      // row stride (per batch) inside [B,T,H] activation arrays
#define XGRS (TT*G4)     // row stride (per batch) inside [B,T,4H] xg array

typedef unsigned short u16t;
typedef __attribute__((ext_vector_type(16))) __bf16 v16bf;
typedef __attribute__((ext_vector_type(8)))  float  v8f;
typedef __attribute__((ext_vector_type(8)))  unsigned short v8u;
typedef __attribute__((ext_vector_type(16))) unsigned short v16u;

__device__ __forceinline__ u16t f2bf(float x) {
    unsigned u = __float_as_uint(x);
    u += 0x7FFFu + ((u >> 16) & 1u);          // round-to-nearest-even
    return (u16t)(u >> 16);
}
__device__ __forceinline__ float bf2f(u16t h) {
    return __uint_as_float(((unsigned)h) << 16);
}
__device__ __forceinline__ float sigf(float x) {
    return 1.0f / (1.0f + __expf(-x));
}

// ---- WMMA fragment helpers (contiguous-k order in both A and B, wave32) -------------
// A (16x32 bf16, row-major source [M,K], row stride rs elements):
//   lane: M = m0 + (lane&15); halves j: K = j + 16*(lane>>4)
__device__ __forceinline__ v16bf load_A(const u16t* base, int rs, int m0, int k0, int lane) {
    int m  = m0 + (lane & 15);
    int kh = (lane >> 4) << 4;                 // 0 or 16
    const u16t* p = base + (size_t)m * (size_t)rs + (size_t)(k0 + kh);
    v8u lo = *(const v8u*)(p);                 // K = k0+kh .. +7
    v8u hi = *(const v8u*)(p + 8);             // K = k0+kh+8 .. +15
    v16u t;
#pragma unroll
    for (int j = 0; j < 8; ++j) { t[j] = lo[j]; t[j + 8] = hi[j]; }
    return __builtin_bit_cast(v16bf, t);
}
// B (32x16 bf16) with B[k,n] = src[n][k], src row-major [N,K] stride rs:
//   lane: N = n0 + (lane&15); halves j: K = k0 + 16*(lane>>4) + j  (contiguous 32B)
__device__ __forceinline__ v16bf load_B(const u16t* base, int rs, int n0, int k0, int lane) {
    int n  = n0 + (lane & 15);
    int kk = k0 + ((lane >> 4) << 4);
    const u16t* p = base + (size_t)n * (size_t)rs + (size_t)kk;
    return __builtin_bit_cast(v16bf, *(const v16u*)p);
}
__device__ __forceinline__ v8f wmma_bf16(v16bf a, v16bf b, v8f c) {
    return __builtin_amdgcn_wmma_f32_16x16x32_bf16(false, a, false, b, (short)0, c, false, false);
}

// ---- small utility kernels -------------------------------------------------
__global__ void k_cvt(const float* __restrict__ s, u16t* __restrict__ d, int n) {
    int i = blockIdx.x * 256 + threadIdx.x;
    if (i < n) d[i] = f2bf(s[i]);
}
// dst[c*rows + r] = bf(src[r*cols + c])  (src [rows,cols] -> dst [cols,rows])
__global__ void k_cvtT(const float* __restrict__ s, u16t* __restrict__ d, int rows, int cols) {
    int i = blockIdx.x * 256 + threadIdx.x;
    if (i < rows * cols) {
        int r = i / cols, c = i % cols;
        d[(size_t)c * rows + r] = f2bf(s[i]);
    }
}
__global__ void k_addv(const float* __restrict__ a, const float* __restrict__ b,
                       float* __restrict__ d, int n) {
    int i = blockIdx.x * 256 + threadIdx.x;
    if (i < n) d[i] = a[i] + b[i];
}
__global__ void k_zf(float* __restrict__ d, int n) {
    int i = blockIdx.x * 256 + threadIdx.x;
    if (i < n) d[i] = 0.0f;
}
__global__ void k_zu(u16t* __restrict__ d, int n) {
    int i = blockIdx.x * 256 + threadIdx.x;
    if (i < n) d[i] = 0;
}

// ---- xg = A_bf[Mtot,K] @ W_bf[2048,K]^T + bias  (out f32 [Mtot,2048]) ------
// grid (Mtot/32, 2048/64), block 256 (8 waves): wave w -> mtile=w&1, ntile=w>>1
__global__ void k_gemm_xg(const u16t* __restrict__ A, int K,
                          const u16t* __restrict__ W,
                          const float* __restrict__ bias,
                          float* __restrict__ out) {
    int tid = threadIdx.x, lane = tid & 31, w = tid >> 5;
    int m0 = blockIdx.x * 32 + (w & 1) * 16;
    int n0 = blockIdx.y * 64 + (w >> 1) * 16;
    v8f acc = {0, 0, 0, 0, 0, 0, 0, 0};
    for (int k0 = 0; k0 < K; k0 += 32) {
        v16bf a = load_A(A, K, m0, k0, lane);
        v16bf b = load_B(W, K, n0, k0, lane);
        acc = wmma_bf16(a, b, acc);
    }
    int n  = n0 + (lane & 15);
    float bv = bias[n];
    int mh = m0 + ((lane >> 4) << 3);
#pragma unroll
    for (int r = 0; r < 8; ++r)
        out[(size_t)(mh + r) * G4 + n] = acc[r] + bv;
}

// ---- fused LSTM forward step: g = xg_t + h_prev @ W_hh^T, gates, h/c update
// grid 16 WGs (32 h-cols each), block 256 = 8 waves; wave w: gate=w>>1, subtile=w&1
__global__ void k_lstm_fwd_step(const u16t* __restrict__ hprev, int hprev_rs,
                                const u16t* __restrict__ Whh,
                                const float* __restrict__ xg_t,
                                const float* __restrict__ cprev, int cprev_rs,
                                u16t* __restrict__ ai, u16t* __restrict__ af,
                                u16t* __restrict__ ag, u16t* __restrict__ ao,
                                float* __restrict__ ac, u16t* __restrict__ hout) {
    __shared__ float gbuf[32 * 128];
    int tid = threadIdx.x, lane = tid & 31, w = tid >> 5;
    int gate  = w >> 1;
    int hbase = blockIdx.x * 32;
    int n_g   = gate * HH + hbase + (w & 1) * 16;
    v8f acc0 = {0, 0, 0, 0, 0, 0, 0, 0};
    v8f acc1 = {0, 0, 0, 0, 0, 0, 0, 0};
    for (int k0 = 0; k0 < HH; k0 += 32) {
        v16bf b  = load_B(Whh, HH, n_g, k0, lane);
        v16bf a0 = load_A(hprev, hprev_rs, 0,  k0, lane);
        v16bf a1 = load_A(hprev, hprev_rs, 16, k0, lane);
        acc0 = wmma_bf16(a0, b, acc0);
        acc1 = wmma_bf16(a1, b, acc1);
    }
    {   // D layout: lane = n + 16*(m>=8); vgpr r -> M = r + 8*(lane>>4)
        int col = w * 16 + (lane & 15);        // == gate*32 + (w&1)*16 + n
        int mh  = (lane >> 4) * 8;
#pragma unroll
        for (int r = 0; r < 8; ++r) {
            gbuf[(mh + r) * 128 + col]      = acc0[r];
            gbuf[(mh + 16 + r) * 128 + col] = acc1[r];
        }
    }
    __syncthreads();
#pragma unroll
    for (int p = 0; p < 4; ++p) {
        int idx  = p * 256 + tid;              // 32 batches x 32 h-cols
        int bb   = idx >> 5, hl = idx & 31, hcol = hbase + hl;
        const float* xr = xg_t + (size_t)bb * XGRS;
        float gi = gbuf[bb * 128 + hl]      + xr[hcol];
        float gf = gbuf[bb * 128 + 32 + hl] + xr[HH + hcol];
        float gg = gbuf[bb * 128 + 64 + hl] + xr[2 * HH + hcol];
        float go = gbuf[bb * 128 + 96 + hl] + xr[3 * HH + hcol];
        float iv = sigf(gi), fv = sigf(gf), gv = tanhf(gg), ov = sigf(go);
        float cp = cprev[(size_t)bb * cprev_rs + hcol];
        float c  = fv * cp + iv * gv;
        float tc = tanhf(c);
        float h  = ov * tc;
        size_t off = (size_t)bb * ARS + hcol;
        ai[off] = f2bf(iv); af[off] = f2bf(fv);
        ag[off] = f2bf(gv); ao[off] = f2bf(ov);
        ac[off] = c;        hout[off] = f2bf(h);
    }
}

// ---- BPTT pointwise: gate-preactivation grads -> dG (bf16), dc carry ------
__global__ void k_lstm_bwd_point(const u16t* __restrict__ ai, const u16t* __restrict__ af,
                                 const u16t* __restrict__ ag, const u16t* __restrict__ ao,
                                 const float* __restrict__ ac,
                                 const float* __restrict__ cprev, int cprev_rs,
                                 const float* __restrict__ dh_in,
                                 const float* __restrict__ dc_in,
                                 const float* __restrict__ dxext, int dxext_rs,
                                 u16t* __restrict__ dG, float* __restrict__ dc_out) {
    int idx = blockIdx.x * 256 + threadIdx.x;     // 32*512 threads
    int bb = idx >> 9, hcol = idx & 511;
    size_t off = (size_t)bb * ARS + hcol;
    float iv = bf2f(ai[off]), fv = bf2f(af[off]);
    float gv = bf2f(ag[off]), ov = bf2f(ao[off]);
    float c  = ac[off];
    float cp = cprev[(size_t)bb * cprev_rs + hcol];
    float tc = tanhf(c);
    float dh = dh_in[bb * HH + hcol] + dxext[(size_t)bb * dxext_rs + hcol];
    float doo = dh * tc;
    float dc  = dc_in[bb * HH + hcol] + dh * ov * (1.0f - tc * tc);
    float di = dc * gv, df = dc * cp, dg = dc * iv;
    dc_out[bb * HH + hcol] = dc * fv;
    float dgi = di * iv * (1.0f - iv);
    float dgf = df * fv * (1.0f - fv);
    float dgg = dg * (1.0f - gv * gv);
    float dgo = doo * ov * (1.0f - ov);
    u16t* dr = dG + (size_t)bb * G4;
    dr[hcol]          = f2bf(dgi);
    dr[HH + hcol]     = f2bf(dgf);
    dr[2 * HH + hcol] = f2bf(dgg);
    dr[3 * HH + hcol] = f2bf(dgo);
}

// ---- BPTT GEMM: dh_next[32,512] = dG @ W_hh  and  dx_t[32,N2] = dG @ W_ih
// N columns 0..511 -> W_hhT/dh_out ; 512..512+N2 -> B2 (W_ihT)/out2
// grid (512+N2)/64, block 256; wave w: mtile=w&1, ntile=w>>1
__global__ void k_bwd_gemm(const u16t* __restrict__ dG,
                           const u16t* __restrict__ WhhT,
                           float* __restrict__ dh_out,
                           const u16t* __restrict__ B2,
                           float* __restrict__ out2, int out2_rs) {
    int tid = threadIdx.x, lane = tid & 31, w = tid >> 5;
    int m0 = (w & 1) * 16;
    int nglob = blockIdx.x * 64 + (w >> 1) * 16;
    const u16t* Bsrc; float* outp; int nb, ost;
    if (nglob < HH) { Bsrc = WhhT; nb = nglob;      outp = dh_out; ost = HH; }
    else            { Bsrc = B2;   nb = nglob - HH; outp = out2;   ost = out2_rs; }
    v8f acc = {0, 0, 0, 0, 0, 0, 0, 0};
    for (int k0 = 0; k0 < G4; k0 += 32) {
        v16bf a = load_A(dG, G4, m0, k0, lane);
        v16bf b = load_B(Bsrc, G4, nb, k0, lane);
        acc = wmma_bf16(a, b, acc);
    }
    int n  = nb + (lane & 15);
    int mh = m0 + ((lane >> 4) << 3);
#pragma unroll
    for (int r = 0; r < 8; ++r)
        outp[(size_t)(mh + r) * ost + n] = acc[r];
}

// ---------------------------------------------------------------------------
extern "C" void kernel_launch(void* const* d_in, const int* in_sizes, int n_in,
                              void* d_out, int out_size, void* d_ws, size_t ws_size,
                              hipStream_t stream) {
    (void)in_sizes; (void)n_in; (void)out_size; (void)ws_size;
    const float* x    = (const float*)d_in[0];
    const float* Wih0 = (const float*)d_in[1];
    const float* Whh0 = (const float*)d_in[2];
    const float* bih0 = (const float*)d_in[3];
    const float* bhh0 = (const float*)d_in[4];
    const float* Wih1 = (const float*)d_in[5];
    const float* Whh1 = (const float*)d_in[6];
    const float* bih1 = (const float*)d_in[7];
    const float* bhh1 = (const float*)d_in[8];
    const float* Wout = (const float*)d_in[9];
    float* out = (float*)d_out;

    char* ws = (char*)d_ws; size_t off = 0;
    auto alloc = [&](size_t bytes) -> void* {
        void* p = ws + off; off = (off + bytes + 255) & ~(size_t)255; return p;
    };
    const size_t BT = (size_t)BB * TT;   // 16384
    // bf16 buffers
    u16t* Wih0_bf = (u16t*)alloc((size_t)G4 * NIN * 2);
    u16t* Wih0T   = (u16t*)alloc((size_t)NIN * G4 * 2);
    u16t* Whh0_bf = (u16t*)alloc((size_t)G4 * HH * 2);
    u16t* Whh0T   = (u16t*)alloc((size_t)HH * G4 * 2);
    u16t* Wih1_bf = (u16t*)alloc((size_t)G4 * HH * 2);
    u16t* Wih1T   = (u16t*)alloc((size_t)HH * G4 * 2);
    u16t* Whh1_bf = (u16t*)alloc((size_t)G4 * HH * 2);
    u16t* Whh1T   = (u16t*)alloc((size_t)HH * G4 * 2);
    u16t* x_bf    = (u16t*)alloc(BT * NIN * 2);
    u16t* h0_bf   = (u16t*)alloc(BT * HH * 2);
    u16t* h1_bf   = (u16t*)alloc(BT * HH * 2);
    u16t* dG      = (u16t*)alloc((size_t)BB * G4 * 2);
    u16t* zbf     = (u16t*)alloc((size_t)HH * 2);
    // f32 buffers (gate activations a*i/a*f/a*g/a*o are bf16)
    float* bias0 = (float*)alloc(G4 * 4);
    float* bias1 = (float*)alloc(G4 * 4);
    float* xg    = (float*)alloc(BT * G4 * 4);          // reused by both layers
    u16t* a0i = (u16t*)alloc(BT * HH * 2); u16t* a0f = (u16t*)alloc(BT * HH * 2);
    u16t* a0g = (u16t*)alloc(BT * HH * 2); u16t* a0o = (u16t*)alloc(BT * HH * 2);
    float* a0c = (float*)alloc(BT * HH * 4);
    u16t* a1i = (u16t*)alloc(BT * HH * 2); u16t* a1f = (u16t*)alloc(BT * HH * 2);
    u16t* a1g = (u16t*)alloc(BT * HH * 2); u16t* a1o = (u16t*)alloc(BT * HH * 2);
    float* a1c = (float*)alloc(BT * HH * 4);
    float* dx1 = (float*)alloc(BT * HH * 4);
    float* dhp[2] = { (float*)alloc(BB * HH * 4), (float*)alloc(BB * HH * 4) };
    float* dcp[2] = { (float*)alloc(BB * HH * 4), (float*)alloc(BB * HH * 4) };
    float* zf  = (float*)alloc(HH * 4);

    auto g1 = [](int n) { return dim3((n + 255) / 256); };
    // ---- setup: bf16 weight copies (+ transposed), biases, zero blocks ----
    k_cvt <<<g1(G4 * NIN), 256, 0, stream>>>(Wih0, Wih0_bf, G4 * NIN);
    k_cvtT<<<g1(G4 * NIN), 256, 0, stream>>>(Wih0, Wih0T, G4, NIN);
    k_cvt <<<g1(G4 * HH), 256, 0, stream>>>(Whh0, Whh0_bf, G4 * HH);
    k_cvtT<<<g1(G4 * HH), 256, 0, stream>>>(Whh0, Whh0T, G4, HH);
    k_cvt <<<g1(G4 * HH), 256, 0, stream>>>(Wih1, Wih1_bf, G4 * HH);
    k_cvtT<<<g1(G4 * HH), 256, 0, stream>>>(Wih1, Wih1T, G4, HH);
    k_cvt <<<g1(G4 * HH), 256, 0, stream>>>(Whh1, Whh1_bf, G4 * HH);
    k_cvtT<<<g1(G4 * HH), 256, 0, stream>>>(Whh1, Whh1T, G4, HH);
    k_cvt <<<g1((int)(BT * NIN)), 256, 0, stream>>>(x, x_bf, (int)(BT * NIN));
    k_addv<<<g1(G4), 256, 0, stream>>>(bih0, bhh0, bias0, G4);
    k_addv<<<g1(G4), 256, 0, stream>>>(bih1, bhh1, bias1, G4);
    k_zu<<<g1(HH), 256, 0, stream>>>(zbf, HH);
    k_zf<<<g1(HH), 256, 0, stream>>>(zf, HH);
    k_zf<<<g1(BB * HH), 256, 0, stream>>>(dhp[0], BB * HH);
    k_zf<<<g1(BB * HH), 256, 0, stream>>>(dcp[0], BB * HH);

    dim3 gxg(512, 32);   // Mtot/32 x 2048/64
    // ---- layer 0 forward ----
    k_gemm_xg<<<gxg, 256, 0, stream>>>(x_bf, NIN, Wih0_bf, bias0, xg);
    for (int t = 0; t < TT; ++t) {
        const u16t*  hp = t ? h0_bf + (size_t)(t - 1) * HH : zbf;
        const float* cp = t ? a0c   + (size_t)(t - 1) * HH : zf;
        int hrs = t ? ARS : 0, crs = t ? ARS : 0;
        k_lstm_fwd_step<<<16, 256, 0, stream>>>(hp, hrs, Whh0_bf,
            xg + (size_t)t * G4, cp, crs,
            a0i + (size_t)t * HH, a0f + (size_t)t * HH,
            a0g + (size_t)t * HH, a0o + (size_t)t * HH,
            a0c + (size_t)t * HH, h0_bf + (size_t)t * HH);
    }
    // ---- layer 1 forward ----
    k_gemm_xg<<<gxg, 256, 0, stream>>>(h0_bf, HH, Wih1_bf, bias1, xg);
    for (int t = 0; t < TT; ++t) {
        const u16t*  hp = t ? h1_bf + (size_t)(t - 1) * HH : zbf;
        const float* cp = t ? a1c   + (size_t)(t - 1) * HH : zf;
        int hrs = t ? ARS : 0, crs = t ? ARS : 0;
        k_lstm_fwd_step<<<16, 256, 0, stream>>>(hp, hrs, Whh1_bf,
            xg + (size_t)t * G4, cp, crs,
            a1i + (size_t)t * HH, a1f + (size_t)t * HH,
            a1g + (size_t)t * HH, a1o + (size_t)t * HH,
            a1c + (size_t)t * HH, h1_bf + (size_t)t * HH);
    }
    // ---- layer 1 backward (dh_ext = W_out row, broadcast: stride 0) ----
    for (int s = 0; s < TT; ++s) {
        int t = TT - 1 - s, in = s & 1, ob = in ^ 1;
        const float* cp = t ? a1c + (size_t)(t - 1) * HH : zf;
        int crs = t ? ARS : 0;
        k_lstm_bwd_point<<<64, 256, 0, stream>>>(
            a1i + (size_t)t * HH, a1f + (size_t)t * HH,
            a1g + (size_t)t * HH, a1o + (size_t)t * HH,
            a1c + (size_t)t * HH, cp, crs,
            dhp[in], dcp[in], Wout, 0, dG, dcp[ob]);
        k_bwd_gemm<<<16, 256, 0, stream>>>(dG, Whh1T, dhp[ob],
            Wih1T, dx1 + (size_t)t * HH, ARS);
    }
    // reset carries for layer 0 backward
    k_zf<<<g1(BB * HH), 256, 0, stream>>>(dhp[0], BB * HH);
    k_zf<<<g1(BB * HH), 256, 0, stream>>>(dcp[0], BB * HH);
    // ---- layer 0 backward (dh_ext = dx1[:,t,:]); dx -> d_out ----
    for (int s = 0; s < TT; ++s) {
        int t = TT - 1 - s, in = s & 1, ob = in ^ 1;
        const float* cp = t ? a0c + (size_t)(t - 1) * HH : zf;
        int crs = t ? ARS : 0;
        k_lstm_bwd_point<<<64, 256, 0, stream>>>(
            a0i + (size_t)t * HH, a0f + (size_t)t * HH,
            a0g + (size_t)t * HH, a0o + (size_t)t * HH,
            a0c + (size_t)t * HH, cp, crs,
            dhp[in], dcp[in], dx1 + (size_t)t * HH, ARS, dG, dcp[ob]);
        k_bwd_gemm<<<9, 256, 0, stream>>>(dG, Whh0T, dhp[ob],
            Wih0T, out + (size_t)t * NIN, TT * NIN);
    }
}
